// MultiHeadSelfAttention_31301721653745
// MI455X (gfx1250) — matrix operand order repaired and checked by reading, not compiled, not run on hardware
//
#include <hip/hip_runtime.h>

typedef _Float16 h16;
typedef __attribute__((ext_vector_type(16))) _Float16 v16h;
typedef __attribute__((ext_vector_type(8)))  _Float16 v8h;
typedef __attribute__((ext_vector_type(8)))  float    v8f;

// ---------------------------------------------------------------------------
// WMMA fragment helpers (layouts per CDNA5 ISA 7.12.2, wave32)
// ---------------------------------------------------------------------------
static __device__ __forceinline__ v16h mk16(v8h lo, v8h hi) {
  return __builtin_shufflevector(lo, hi, 0,1,2,3,4,5,6,7,8,9,10,11,12,13,14,15);
}

// A-matrix 16x32 f16: lane holds row (lane&15); k-chunks [koff..koff+7] and
// [koff+16..koff+23] with koff = (lane<16)?0:8.  base -> (row0, k0).
static __device__ __forceinline__ v16h load_a(const h16* base, int lda, int lane) {
  const int r    = lane & 15;
  const int koff = (lane < 16) ? 0 : 8;
  const h16* p   = base + (size_t)r * lda + koff;
  v8h lo = *(const v8h*)(p);
  v8h hi = *(const v8h*)(p + 16);
  return mk16(lo, hi);
}

// B-matrix 32x16 f16: lane holds column (lane&15); column c of B is row c of a
// row-major [16 x 32] source; k-chunks [koff..koff+7] and [koff+16..koff+23]
// with koff = (lane<16)?0:8.
static __device__ __forceinline__ v16h load_b(const h16* base, int ldb, int lane) {
  const int c    = lane & 15;
  const int koff = (lane < 16) ? 0 : 8;
  const h16* p   = base + (size_t)c * ldb + koff;
  v8h lo = *(const v8h*)(p);
  v8h hi = *(const v8h*)(p + 16);
  return mk16(lo, hi);
}

static __device__ __forceinline__ v8f wmma16(v16h a, v16h b, v8f c) {
  return __builtin_amdgcn_wmma_f32_16x16x32_f16(false, a, false, b, (short)0, c,
                                                false, false);
}

// ---------------------------------------------------------------------------
// Kernel 0: f32 -> f16 convert
// ---------------------------------------------------------------------------
__global__ void cvt_f32_f16(const float* __restrict__ in, h16* __restrict__ out, int n) {
  int i = blockIdx.x * blockDim.x + threadIdx.x;
  if (i < n) out[i] = (h16)in[i];
}

// ---------------------------------------------------------------------------
// Kernel 1: QKV projection.  C[8192,3072] = X[8192,1024] @ Wqkv^T + bqkv,
// scattered into Q[bh,n,d], K[bh,n,d], Vt[bh,d,n] (f16).
// Block = 256 threads = 8 waves (2 M x 4 N); wave tile 32x64; block 64x256.
// ---------------------------------------------------------------------------
__global__ __launch_bounds__(256) void gemm_qkv(
    const h16* __restrict__ X, const h16* __restrict__ W,
    const float* __restrict__ bias,
    h16* __restrict__ Qb, h16* __restrict__ Kb, h16* __restrict__ Vt) {
  const int lane = threadIdx.x & 31, wid = threadIdx.x >> 5;
  const int m0 = blockIdx.y * 64 + (wid & 1) * 32;
  const int n0 = blockIdx.x * 256 + (wid >> 1) * 64;

  v8f acc[2][4];
#pragma unroll
  for (int a = 0; a < 2; a++)
#pragma unroll
    for (int b = 0; b < 4; b++)
#pragma unroll
      for (int i = 0; i < 8; i++) acc[a][b][i] = 0.f;

  for (int k0 = 0; k0 < 1024; k0 += 32) {
    v16h a0 = load_a(X + (size_t)m0 * 1024 + k0, 1024, lane);
    v16h a1 = load_a(X + (size_t)(m0 + 16) * 1024 + k0, 1024, lane);
#pragma unroll
    for (int tn = 0; tn < 4; tn++) {
      v16h bf = load_b(W + (size_t)(n0 + tn * 16) * 1024 + k0, 1024, lane);
      acc[0][tn] = wmma16(a0, bf, acc[0][tn]);
      acc[1][tn] = wmma16(a1, bf, acc[1][tn]);
    }
  }

  const int c = lane & 15, rb = (lane < 16) ? 0 : 8;
#pragma unroll
  for (int tm = 0; tm < 2; tm++) {
#pragma unroll
    for (int tn = 0; tn < 4; tn++) {
      const int o = n0 + tn * 16 + c;          // 0..3071
      const int hh = o / 192, rem = o % 192;   // head, within-head
      const int t = rem / 64, d = rem % 64;    // 0=q,1=k,2=v
      const float bv = bias[o];
#pragma unroll
      for (int i = 0; i < 8; i++) {
        const int m = m0 + tm * 16 + rb + i;   // 0..8191 = b*2048 + nseq
        const int bb = m >> 11, ns = m & 2047;
        const h16 val = (h16)(acc[tm][tn][i] + bv);
        const size_t bh = (size_t)(bb * 16 + hh);
        if (t == 0)      Qb[(bh * 2048 + ns) * 64 + d] = val;
        else if (t == 1) Kb[(bh * 2048 + ns) * 64 + d] = val;
        else             Vt[(bh * 64 + d) * 2048 + ns] = val;
      }
    }
  }
}

// ---------------------------------------------------------------------------
// Kernel 2: fused flash attention per (b,h).  Workgroup = 128 q-rows (8 waves
// x 16 rows).  K/V tiles (32 keys) staged ONCE per workgroup into LDS with
// async global->LDS b128 copies (ASYNCcnt), double buffered so the next
// tile's DMA overlaps this tile's 8 WMMAs + online softmax.
// ---------------------------------------------------------------------------
__global__ __launch_bounds__(256) void attn_fused(
    const h16* __restrict__ Qb, const h16* __restrict__ Kb,
    const h16* __restrict__ Vt, h16* __restrict__ Ab) {
  // LDS layout (halves):
  //   K bufs : 2 x [32 keys][72]  @ 0     (4608)   row stride 72 -> conflict-free
  //   V bufs : 2 x [64 d  ][40]   @ 4608  (5120)   row stride 40 -> conflict-free
  //   P tile : 8 waves x [16][40] @ 9728  (5120)
  __shared__ __attribute__((aligned(16))) h16 smem[14848];

  const int lane = threadIdx.x & 31, wid = threadIdx.x >> 5;
  const int bh = blockIdx.y;
  const int b = bh >> 4, hh = bh & 15;
  const int q0 = blockIdx.x * 128 + wid * 16;

  const h16* Qp = Qb + (size_t)bh * 2048 * 64;
  const h16* Kp = Kb + (size_t)bh * 2048 * 64;
  const h16* Vp = Vt + (size_t)bh * 64 * 2048;
  h16* lp = smem + 9728 + wid * 16 * 40;

  // cooperative async copy of one 32-key K/V tile: 256 threads x 16B each side
  const int tid  = threadIdx.x;
  const int krow = tid >> 3, kseg = tid & 7;   // K: 32 rows x 8 chunks of 8 halves
  const int vrow = tid >> 2, vseg = tid & 3;   // V: 64 rows x 4 chunks of 8 halves
  auto copy_tile = [&](int kb, int bi) {
    const int kbaseH = bi ? 2304 : 0;                 // half-element offsets
    const int vbaseH = 4608 + (bi ? 2560 : 0);
    const h16* gK = Kp + (size_t)(kb + krow) * 64 + kseg * 8;
    const h16* gV = Vp + (size_t)vrow * 2048 + kb + vseg * 8;
    h16* lK = smem + kbaseH + krow * 72 + kseg * 8;   // runtime addrspacecast
    h16* lV = smem + vbaseH + vrow * 40 + vseg * 8;
    unsigned offK = (unsigned)(unsigned long long)(void*)lK;
    unsigned offV = (unsigned)(unsigned long long)(void*)lV;
    asm volatile("global_load_async_to_lds_b128 %0, %1, off"
                 :: "v"(offK), "v"(gK) : "memory");
    asm volatile("global_load_async_to_lds_b128 %0, %1, off"
                 :: "v"(offV), "v"(gV) : "memory");
  };

  copy_tile(0, 0);

  // Q fragments for this wave's 16 rows (dh = 64 -> two 16x32 A frags)
  const v16h aQ0 = load_a(Qp + (size_t)q0 * 64 + 0, 64, lane);
  const v16h aQ1 = load_a(Qp + (size_t)q0 * 64 + 32, 64, lane);

  v8f o0, o1, o2, o3;
  float rmax[8], rsum[8];
#pragma unroll
  for (int i = 0; i < 8; i++) {
    o0[i] = o1[i] = o2[i] = o3[i] = 0.f;
    rmax[i] = -3.0e38f;
    rsum[i] = 0.f;
  }

  const int c = lane & 15, rb = (lane < 16) ? 0 : 8;

  int p = 0;
  for (int kb = 0; kb < 2048; kb += 32) {
    asm volatile("s_wait_asynccnt 0x0" ::: "memory");  // this wave's DMA done
    __syncthreads();                                   // tile p visible to all
    if (kb + 32 < 2048) copy_tile(kb + 32, p ^ 1);     // overlap next DMA

    const h16* Kt = smem + (p ? 2304 : 0);             // [32 keys][72]
    const h16* Vl = smem + 4608 + (p ? 2560 : 0);      // [64 d][40]

    // ---- scores: two 16x16 tiles over local keys [0..15], [16..31]
    v8f s0 = {0.f,0.f,0.f,0.f,0.f,0.f,0.f,0.f};
    v8f s1 = {0.f,0.f,0.f,0.f,0.f,0.f,0.f,0.f};
    s0 = wmma16(aQ0, load_b(Kt + 0 * 72 + 0,  72, lane), s0);
    s0 = wmma16(aQ1, load_b(Kt + 0 * 72 + 32, 72, lane), s0);
    s1 = wmma16(aQ0, load_b(Kt + 16 * 72 + 0,  72, lane), s1);
    s1 = wmma16(aQ1, load_b(Kt + 16 * 72 + 32, 72, lane), s1);

    // ---- online softmax (scale 1/sqrt(64) = 0.125), rows across 16 lanes
#pragma unroll
    for (int i = 0; i < 8; i++) {
      s0[i] *= 0.125f;
      s1[i] *= 0.125f;
      float tmax = fmaxf(s0[i], s1[i]);
      tmax = fmaxf(tmax, __shfl_xor(tmax, 1, 16));
      tmax = fmaxf(tmax, __shfl_xor(tmax, 2, 16));
      tmax = fmaxf(tmax, __shfl_xor(tmax, 4, 16));
      tmax = fmaxf(tmax, __shfl_xor(tmax, 8, 16));
      const float mnew = fmaxf(rmax[i], tmax);
      const float alpha = __expf(rmax[i] - mnew);
      rmax[i] = mnew;
      s0[i] = __expf(s0[i] - mnew);
      s1[i] = __expf(s1[i] - mnew);
      float ps = s0[i] + s1[i];
      ps += __shfl_xor(ps, 1, 16);
      ps += __shfl_xor(ps, 2, 16);
      ps += __shfl_xor(ps, 4, 16);
      ps += __shfl_xor(ps, 8, 16);
      rsum[i] = rsum[i] * alpha + ps;
      o0[i] *= alpha; o1[i] *= alpha; o2[i] *= alpha; o3[i] *= alpha;
    }

    // ---- transpose P (D-layout -> A-layout) through this wave's LDS tile
#pragma unroll
    for (int i = 0; i < 8; i++) {
      lp[(rb + i) * 40 + c]      = (h16)s0[i];
      lp[(rb + i) * 40 + 16 + c] = (h16)s1[i];
    }
    asm volatile("s_wait_dscnt 0x0" ::: "memory");
    const v16h aP = load_a(lp, 40, lane);   // 16 rows x 32 keys

    // ---- PV from LDS V tile
    o0 = wmma16(aP, load_b(Vl + 0 * 40,  40, lane), o0);
    o1 = wmma16(aP, load_b(Vl + 16 * 40, 40, lane), o1);
    o2 = wmma16(aP, load_b(Vl + 32 * 40, 40, lane), o2);
    o3 = wmma16(aP, load_b(Vl + 48 * 40, 40, lane), o3);

    p ^= 1;
  }

  // ---- normalize and emit f16 rows into [b, n, h*64 + d]
#pragma unroll
  for (int i = 0; i < 8; i++) {
    const float inv = 1.f / rsum[i];
    const int q = q0 + rb + i;
    h16* outp = Ab + ((size_t)b * 2048 + q) * 1024 + hh * 64;
    outp[0  + c] = (h16)(o0[i] * inv);
    outp[16 + c] = (h16)(o1[i] * inv);
    outp[32 + c] = (h16)(o2[i] * inv);
    outp[48 + c] = (h16)(o3[i] * inv);
  }
}

// ---------------------------------------------------------------------------
// Kernel 3: output projection.  Out[8192,1024] = A @ Wo^T + bo  (f32 out)
// Wave tile 32x64; block 64x256.
// ---------------------------------------------------------------------------
__global__ __launch_bounds__(256) void gemm_out(
    const h16* __restrict__ A, const h16* __restrict__ W,
    const float* __restrict__ bias, float* __restrict__ Out) {
  const int lane = threadIdx.x & 31, wid = threadIdx.x >> 5;
  const int m0 = blockIdx.y * 64 + (wid & 1) * 32;
  const int n0 = blockIdx.x * 256 + (wid >> 1) * 64;

  v8f acc[2][4];
#pragma unroll
  for (int a = 0; a < 2; a++)
#pragma unroll
    for (int b = 0; b < 4; b++)
#pragma unroll
      for (int i = 0; i < 8; i++) acc[a][b][i] = 0.f;

  for (int k0 = 0; k0 < 1024; k0 += 32) {
    v16h a0 = load_a(A + (size_t)m0 * 1024 + k0, 1024, lane);
    v16h a1 = load_a(A + (size_t)(m0 + 16) * 1024 + k0, 1024, lane);
#pragma unroll
    for (int tn = 0; tn < 4; tn++) {
      v16h bf = load_b(W + (size_t)(n0 + tn * 16) * 1024 + k0, 1024, lane);
      acc[0][tn] = wmma16(a0, bf, acc[0][tn]);
      acc[1][tn] = wmma16(a1, bf, acc[1][tn]);
    }
  }

  const int c = lane & 15, rb = (lane < 16) ? 0 : 8;
#pragma unroll
  for (int tm = 0; tm < 2; tm++) {
#pragma unroll
    for (int tn = 0; tn < 4; tn++) {
      const int n = n0 + tn * 16 + c;
      const float bv = bias[n];
#pragma unroll
      for (int i = 0; i < 8; i++) {
        const int m = m0 + tm * 16 + rb + i;
        Out[(size_t)m * 1024 + n] = acc[tm][tn][i] + bv;
      }
    }
  }
}

// ---------------------------------------------------------------------------
// Host launcher
// ---------------------------------------------------------------------------
extern "C" void kernel_launch(void* const* d_in, const int* in_sizes, int n_in,
                              void* d_out, int out_size, void* d_ws, size_t ws_size,
                              hipStream_t stream) {
  (void)in_sizes; (void)n_in; (void)out_size; (void)ws_size;
  const float* x    = (const float*)d_in[0];   // [4,2048,1024]
  const float* Wqkv = (const float*)d_in[1];   // [3072,1024]
  const float* bqkv = (const float*)d_in[2];   // [3072]
  const float* Wo   = (const float*)d_in[3];   // [1024,1024]
  const float* bo   = (const float*)d_in[4];   // [1024]
  float* out = (float*)d_out;                  // [4,2048,1024] f32

  char* ws = (char*)d_ws;
  h16* x16    = (h16*)(ws + 0);          // 16 MB  (8192*1024)
  h16* wqkv16 = (h16*)(ws + 16777216);   //  6 MB  (3072*1024)
  h16* wo16   = (h16*)(ws + 23068672);   //  2 MB  (1024*1024)
  h16* qbuf   = (h16*)(ws + 25165824);   // 16 MB  [bh, n, 64]
  h16* kbuf   = (h16*)(ws + 41943040);   // 16 MB  [bh, n, 64]
  h16* vtbuf  = (h16*)(ws + 58720256);   // 16 MB  [bh, 64, n]
  h16* abuf   = (h16*)(ws + 75497472);   // 16 MB  [8192, 1024]

  const int nx = 8192 * 1024, nwq = 3072 * 1024, nwo = 1024 * 1024;
  cvt_f32_f16<<<(nx  + 255) / 256, 256, 0, stream>>>(x,    x16,    nx);
  cvt_f32_f16<<<(nwq + 255) / 256, 256, 0, stream>>>(Wqkv, wqkv16, nwq);
  cvt_f32_f16<<<(nwo + 255) / 256, 256, 0, stream>>>(Wo,   wo16,   nwo);

  gemm_qkv  <<<dim3(12, 128), 256, 0, stream>>>(x16, wqkv16, bqkv, qbuf, kbuf, vtbuf);
  attn_fused<<<dim3(16, 64),  256, 0, stream>>>(qbuf, kbuf, vtbuf, abuf);
  gemm_out  <<<dim3(4, 128),  256, 0, stream>>>(abuf, wo16, bo, out);
}
